// LocalAttention_46806553592394
// MI455X (gfx1250) — matrix operand order repaired and checked by reading, not compiled, not run on hardware
//
#include <hip/hip_runtime.h>

typedef __attribute__((ext_vector_type(16))) __bf16 v16bf;
typedef __attribute__((ext_vector_type(8)))  float  v8f;

struct alignas(16) U4 { unsigned int a, b, c, d; };

union FragBF {
    v16bf v;
    U4 q[2];
    unsigned short h[16];
};

__device__ __forceinline__ unsigned short f2bf(float f) {
    unsigned int u = __float_as_uint(f);
    unsigned int r = u + 0x7FFFu + ((u >> 16) & 1u);
    return (unsigned short)(r >> 16);
}

__device__ __forceinline__ v8f wmma_bf(const FragBF& a, const FragBF& b, v8f c) {
    return __builtin_amdgcn_wmma_f32_16x16x32_bf16(
        false, a.v, false, b.v, (short)0, c, false, false);
}

// Async DMA: global -> LDS, 16 bytes per lane, tracked by ASYNCcnt.
__device__ __forceinline__ void async_g2l_b128(const void* gsrc, void* ldst) {
    unsigned ldsa = (unsigned)(size_t)ldst;          // low 32 bits = LDS byte offset
    unsigned long long ga = (unsigned long long)(size_t)gsrc;
    asm volatile("global_load_async_to_lds_b128 %0, %1, off"
                 :: "v"(ldsa), "v"(ga) : "memory");
}
__device__ __forceinline__ void wait_async0() {
    asm volatile("s_wait_asynccnt 0" ::: "memory");
}

// ---------------------------------------------------------------- cast f32->bf16
__global__ void cast_kernel(const float* __restrict__ in,
                            unsigned short* __restrict__ out, int n) {
    int i = blockIdx.x * blockDim.x + threadIdx.x;
    if (i < n) out[i] = f2bf(in[i]);
}

// ---------------------------------------------------------------- LayerNorm (scale only), fp32 stats -> bf16
__global__ void ln_kernel(const float* __restrict__ x,
                          const float* __restrict__ gamma,
                          unsigned short* __restrict__ xn) {
    const int DIM = 1024;
    int row = blockIdx.x;
    const float4* xr = (const float4*)(x + (size_t)row * DIM);
    float4 v = xr[threadIdx.x];
    float s  = v.x + v.y + v.z + v.w;
    float s2 = v.x * v.x + v.y * v.y + v.z * v.z + v.w * v.w;
    for (int off = 1; off < 32; off <<= 1) {
        s  += __shfl_xor(s,  off, 32);
        s2 += __shfl_xor(s2, off, 32);
    }
    __shared__ float red[16];
    int wid = threadIdx.x >> 5, ln = threadIdx.x & 31;
    if (ln == 0) { red[wid] = s; red[8 + wid] = s2; }
    __syncthreads();
    s = 0.f; s2 = 0.f;
    for (int i = 0; i < 8; i++) { s += red[i]; s2 += red[8 + i]; }
    float mean = s * (1.f / 1024.f);
    float var  = s2 * (1.f / 1024.f) - mean * mean;
    float rstd = rsqrtf(var + 1e-5f);
    const float4* g4 = (const float4*)gamma;
    float4 g = g4[threadIdx.x];
    unsigned short* out = xn + (size_t)row * DIM + threadIdx.x * 4;
    out[0] = f2bf((v.x - mean) * rstd * g.x);
    out[1] = f2bf((v.y - mean) * rstd * g.y);
    out[2] = f2bf((v.z - mean) * rstd * g.z);
    out[3] = f2bf((v.w - mean) * rstd * g.w);
}

// ---------------------------------------------------------------- bf16 GEMM, 128x128x32, double-buffered, async A staging
#define BM 128
#define BN 128
#define BK 32

template <bool OUTF32>
__global__ void __launch_bounds__(256) gemm_bf16_kernel(
    const unsigned short* __restrict__ A,   // MxK row-major bf16
    const unsigned short* __restrict__ B,   // KxN row-major bf16
    unsigned short* __restrict__ Cb,        // bf16 out (if !OUTF32)
    float* __restrict__ Cf,                 // f32 out (if OUTF32)
    const float* __restrict__ bias,
    int M, int N, int K) {
    __shared__ __align__(16) unsigned short lA[2][BM * BK];  // [m][k]
    __shared__ __align__(16) unsigned short lB[2][BN * BK];  // transposed: [n][k]

    int t = threadIdx.x;
    int lane = t & 31;
    int wv = t >> 5;
    int wr = wv >> 2;   // 0..1 -> 64 rows each
    int wc = wv & 3;    // 0..3 -> 32 cols each
    int m0 = blockIdx.y * BM;
    int n0 = blockIdx.x * BN;
    int l15 = lane & 15, lh = lane >> 4;

    v8f zero = {};
    v8f acc[4][2];
    for (int mt = 0; mt < 4; mt++)
        for (int nt = 0; nt < 2; nt++) acc[mt][nt] = zero;

    auto stage = [&](int buf, int k0) {
        {   // A tile: async DMA, 1 x b128 per thread
            int row = t >> 1, kc = (t & 1) * 16;
            const unsigned short* src = A + (size_t)(m0 + row) * K + k0 + kc;
            async_g2l_b128(src, &lA[buf][row * BK + kc]);
        }
        {   // B tile transposed: 2 x b128 coalesced reads, scalar LDS scatter
            unsigned short* dst = &lB[buf][0];
            for (int i = 0; i < 2; i++) {
                int u = t * 2 + i;
                int k = u >> 4, n8 = (u & 15) * 8;
                U4 val = *(const U4*)(B + (size_t)(k0 + k) * N + n0 + n8);
                const unsigned short* hv = (const unsigned short*)&val;
                for (int j = 0; j < 8; j++)
                    dst[(n8 + j) * BK + k] = hv[j];
            }
        }
    };

    stage(0, 0);
    wait_async0();
    __syncthreads();

    int nK = K / BK;
    for (int kk = 0; kk < nK; kk++) {
        int cur = kk & 1;
        if (kk + 1 < nK) stage(cur ^ 1, (kk + 1) * BK);

        FragBF af[4], bfv[2];
        const unsigned short* pA = &lA[cur][0];
        const unsigned short* pB = &lB[cur][0];
        for (int mt = 0; mt < 4; mt++) {
            int m = wr * 64 + mt * 16 + l15;
            af[mt].q[0] = *(const U4*)(pA + m * BK + lh * 8);
            af[mt].q[1] = *(const U4*)(pA + m * BK + 16 + lh * 8);
        }
        for (int nt = 0; nt < 2; nt++) {
            int n = wc * 32 + nt * 16 + l15;
            bfv[nt].q[0] = *(const U4*)(pB + n * BK + lh * 8);
            bfv[nt].q[1] = *(const U4*)(pB + n * BK + 16 + lh * 8);
        }
        for (int mt = 0; mt < 4; mt++)
            for (int nt = 0; nt < 2; nt++)
                acc[mt][nt] = wmma_bf(af[mt], bfv[nt], acc[mt][nt]);

        wait_async0();
        __syncthreads();
    }

    // epilogue: C layout row = v + 8*(lane>>4), col = lane&15
    for (int mt = 0; mt < 4; mt++) {
        for (int nt = 0; nt < 2; nt++) {
            for (int v = 0; v < 8; v++) {
                int r = m0 + wr * 64 + mt * 16 + lh * 8 + v;
                int c = n0 + wc * 32 + nt * 16 + l15;
                float val = acc[mt][nt][v];
                if constexpr (OUTF32)
                    Cf[(size_t)r * N + c] = val + bias[c];
                else
                    Cb[(size_t)r * N + c] = f2bf(val);
            }
        }
    }
}

// ---------------------------------------------------------------- windowed flash attention
// grid (32 windows, 16 heads), 512 threads = 16 waves; wave owns 32 query rows.
__global__ void __launch_bounds__(512) attn_kernel(
    const unsigned short* __restrict__ qkv,   // (16384, 3072) bf16: [q|k|v]
    unsigned short* __restrict__ aout) {      // (16384, 1024) bf16
    const int WSZ = 512;
    int w = blockIdx.x, hh = blockIdx.y;
    int t = threadIdx.x, lane = t & 31, wv = t >> 5;
    int l15 = lane & 15, lh = lane >> 4;
    int ibase = wv * 32;

    __shared__ __align__(16) unsigned short lK[32 * 64];    // [key][d]
    __shared__ __align__(16) unsigned short lVt[64 * 32];   // [d][key]
    __shared__ __align__(16) unsigned short lP[16][32 * 32];// per-wave P [m][key]

    // Q fragments resident: 2 M-tiles x 2 K-steps (d 0..31 / 32..63)
    FragBF qf[2][2];
    for (int mt = 0; mt < 2; mt++) {
        size_t row = (size_t)(w * WSZ + ibase + mt * 16 + l15);
        const unsigned short* qr = qkv + row * 3072 + hh * 64;
        for (int ds = 0; ds < 2; ds++) {
            qf[mt][ds].q[0] = *(const U4*)(qr + ds * 32 + lh * 8);
            qf[mt][ds].q[1] = *(const U4*)(qr + ds * 32 + 16 + lh * 8);
        }
    }

    v8f zero = {};
    v8f o[2][4];
    float mrow[2][8], lrow[2][8];
    for (int mt = 0; mt < 2; mt++) {
        for (int nd = 0; nd < 4; nd++) o[mt][nd] = zero;
        for (int v = 0; v < 8; v++) { mrow[mt][v] = -3.0e38f; lrow[mt][v] = 0.f; }
    }
    const float scale = 0.125f;  // 64^-0.5

    for (int c = 0; c < 32; c++) {
        __syncthreads();
        int j0 = c * 32;
        // ---- stage K (threads 0..255, async DMA) and V^T (threads 256..511)
        if (t < 256) {
            int kr = t >> 3, dq = (t & 7) * 8;
            if (w == 0 && c < 16) {
                U4 z = {0, 0, 0, 0};
                *(U4*)(lK + kr * 64 + dq) = z;
            } else {
                int grow = (w - 1) * WSZ + j0 + kr;
                async_g2l_b128(qkv + (size_t)grow * 3072 + 1024 + hh * 64 + dq,
                               lK + kr * 64 + dq);
            }
        } else {
            int t2 = t - 256;
            int kr = t2 >> 3, d0 = (t2 & 7) * 8;
            U4 val = {0, 0, 0, 0};
            if (!(w == 0 && c < 16)) {
                int grow = (w - 1) * WSZ + j0 + kr;
                val = *(const U4*)(qkv + (size_t)grow * 3072 + 2048 + hh * 64 + d0);
            }
            const unsigned short* hv = (const unsigned short*)&val;
            for (int j = 0; j < 8; j++) lVt[(d0 + j) * 32 + kr] = hv[j];
        }
        wait_async0();
        __syncthreads();

        if (j0 > ibase + 543) continue;      // fully masked for this wave
        bool diag = (j0 - ibase) == 512;     // only chunk crossing the causal edge

        // ---- sim = Q @ K^T  (2 M-tiles x 2 key-tiles)
        v8f s[2][2];
        for (int mt = 0; mt < 2; mt++)
            for (int nt = 0; nt < 2; nt++) s[mt][nt] = zero;
        for (int ds = 0; ds < 2; ds++) {
            FragBF kb[2];
            for (int nt = 0; nt < 2; nt++) {
                int key = nt * 16 + l15;
                kb[nt].q[0] = *(const U4*)(lK + key * 64 + ds * 32 + lh * 8);
                kb[nt].q[1] = *(const U4*)(lK + key * 64 + ds * 32 + 16 + lh * 8);
            }
            for (int mt = 0; mt < 2; mt++)
                for (int nt = 0; nt < 2; nt++)
                    s[mt][nt] = wmma_bf(qf[mt][ds], kb[nt], s[mt][nt]);
        }

        // ---- online softmax per M-tile
        for (int mt = 0; mt < 2; mt++) {
            float pm[2][8];
            for (int nt = 0; nt < 2; nt++)
                for (int v = 0; v < 8; v++)
                    pm[nt][v] = s[mt][nt][v] * scale;
            if (diag) {
                for (int nt = 0; nt < 2; nt++) {
                    int jcol = nt * 16 + l15;
                    for (int v = 0; v < 8; v++) {
                        int irow = mt * 16 + lh * 8 + v;
                        if (jcol > irow) pm[nt][v] = -3.0e38f;  // exp underflows to 0
                    }
                }
            }
            for (int v = 0; v < 8; v++) {
                float tv = fmaxf(pm[0][v], pm[1][v]);
                for (int off = 1; off < 16; off <<= 1)
                    tv = fmaxf(tv, __shfl_xor(tv, off, 16));
                float mnew = fmaxf(mrow[mt][v], tv);
                float al = __expf(mrow[mt][v] - mnew);
                float p0 = __expf(pm[0][v] - mnew);
                float p1 = __expf(pm[1][v] - mnew);
                float rs = p0 + p1;
                for (int off = 1; off < 16; off <<= 1)
                    rs += __shfl_xor(rs, off, 16);
                lrow[mt][v] = lrow[mt][v] * al + rs;
                mrow[mt][v] = mnew;
                pm[0][v] = p0; pm[1][v] = p1;
                for (int nd = 0; nd < 4; nd++) o[mt][nd][v] *= al;
            }
            // spill P to per-wave LDS in [m][key] order for A-fragment reload
            unsigned short* Pw = &lP[wv][0];
            for (int nt = 0; nt < 2; nt++) {
                int jl = nt * 16 + l15;
                for (int v = 0; v < 8; v++) {
                    int ml = mt * 16 + lh * 8 + v;
                    Pw[ml * 32 + jl] = f2bf(pm[nt][v]);
                }
            }
        }

        // ---- O += P @ V
        FragBF vb[4];
        for (int nd = 0; nd < 4; nd++) {
            int d = nd * 16 + l15;
            vb[nd].q[0] = *(const U4*)(lVt + d * 32 + lh * 8);
            vb[nd].q[1] = *(const U4*)(lVt + d * 32 + 16 + lh * 8);
        }
        const unsigned short* Pw = &lP[wv][0];
        for (int mt = 0; mt < 2; mt++) {
            FragBF pf;
            int ml = mt * 16 + l15;
            pf.q[0] = *(const U4*)(Pw + ml * 32 + lh * 8);
            pf.q[1] = *(const U4*)(Pw + ml * 32 + 16 + lh * 8);
            for (int nd = 0; nd < 4; nd++)
                o[mt][nd] = wmma_bf(pf, vb[nd], o[mt][nd]);
        }
    }

    // ---- finalize: divide by row sums, write bf16 to (n, h*64+d)
    for (int mt = 0; mt < 2; mt++) {
        for (int v = 0; v < 8; v++) {
            float inv = 1.f / lrow[mt][v];
            size_t r = (size_t)(w * WSZ + ibase + mt * 16 + lh * 8 + v);
            for (int nd = 0; nd < 4; nd++) {
                int cc = hh * 64 + nd * 16 + l15;
                aout[r * 1024 + cc] = f2bf(o[mt][nd][v] * inv);
            }
        }
    }
}

// ---------------------------------------------------------------- host
extern "C" void kernel_launch(void* const* d_in, const int* in_sizes, int n_in,
                              void* d_out, int out_size, void* d_ws, size_t ws_size,
                              hipStream_t stream) {
    (void)in_sizes; (void)n_in; (void)out_size; (void)ws_size;
    const float* x      = (const float*)d_in[0];
    const float* gamma  = (const float*)d_in[1];
    const float* w_qkv  = (const float*)d_in[2];
    const float* w_out  = (const float*)d_in[3];
    const float* b_out  = (const float*)d_in[4];
    float* out = (float*)d_out;

    const int N = 16384, DIM = 1024, IN3 = 3072;

    unsigned short* xn   = (unsigned short*)d_ws;          // 32 MB
    unsigned short* wq16 = xn   + (size_t)N * DIM;         //  6 MB
    unsigned short* wo16 = wq16 + (size_t)DIM * IN3;       //  2 MB
    unsigned short* qkv  = wo16 + (size_t)DIM * DIM;       // 96 MB
    unsigned short* aout = qkv  + (size_t)N * IN3;         // 32 MB

    cast_kernel<<<(DIM * IN3 + 255) / 256, 256, 0, stream>>>(w_qkv, wq16, DIM * IN3);
    cast_kernel<<<(DIM * DIM + 255) / 256, 256, 0, stream>>>(w_out, wo16, DIM * DIM);
    ln_kernel<<<N, 256, 0, stream>>>(x, gamma, xn);
    gemm_bf16_kernel<false><<<dim3(IN3 / BN, N / BM), 256, 0, stream>>>(
        xn, wq16, qkv, nullptr, nullptr, N, IN3, DIM);
    attn_kernel<<<dim3(32, 16), 512, 0, stream>>>(qkv, aout);
    gemm_bf16_kernel<true><<<dim3(DIM / BN, N / BM), 256, 0, stream>>>(
        aout, wo16, nullptr, out, b_out, N, DIM, DIM);
}
